// MultiHead_36558761623692
// MI455X (gfx1250) — matrix operand order repaired and checked by reading, not compiled, not run on hardware
//
#include <hip/hip_runtime.h>
#include <hip/hip_bf16.h>
#include <math.h>

typedef __bf16 bf16_t;
typedef __attribute__((ext_vector_type(2)))  __bf16 v2bf;
typedef __attribute__((ext_vector_type(16))) __bf16 v16bf;
typedef __attribute__((ext_vector_type(8)))  float  v8f;

#define WSZ     7
#define SHIFT_  3
#define HEADS_  8
#define DH      32
#define NTOK    49          // 7*7 tokens per window
#define IMG     112
#define CCH     256
#define NWIN_TOT 2048       // B * (16*16)
#define MROWS   100352      // NWIN_TOT * NTOK

// ---------------------------------------------------------------------------
// WMMA fragment loaders (bf16, 16x16x32), layouts per CDNA5 ISA
// A 16x32: lanes 0-15 rows 0-15 K 0-15; lanes 16-31 same rows K 16-31
// B 32x16: stored transposed in LDS as [col][k]; lanes 0-15 K 0-15, lanes 16-31 K 16-31
// ---------------------------------------------------------------------------
__device__ __forceinline__ v16bf frag_a(const bf16_t* base, int stride, int lane) {
  const int half = lane >> 4;
  const int r    = lane & 15;
  const bf16_t* p = base + r * stride + half * 16;
  v16bf f;
#pragma unroll
  for (int i = 0; i < 8; ++i) f[i] = p[i];
  p += 8;
#pragma unroll
  for (int i = 0; i < 8; ++i) f[8 + i] = p[i];
  return f;
}

__device__ __forceinline__ v16bf frag_b(const bf16_t* baseT, int stride, int lane) {
  const int half = lane >> 4;
  const int c    = lane & 15;
  const bf16_t* p = baseT + c * stride + half * 16;
  v16bf f;
#pragma unroll
  for (int i = 0; i < 16; ++i) f[i] = p[i];
  return f;
}

#define WMMA_BF16(A, B, C) \
  __builtin_amdgcn_wmma_f32_16x16x32_bf16(false, (A), false, (B), (short)0, (C), false, false)

// Pack two fp32 (adjacent K) into one 32-bit bf16 pair store (lo = even K).
__device__ __forceinline__ void pack_store(bf16_t* dst, float lo, float hi) {
  v2bf p; p[0] = (bf16_t)lo; p[1] = (bf16_t)hi;
  *(v2bf*)dst = p;
}

// ---------------------------------------------------------------------------
// Kernel 1: QKV GEMM  [100352 x 256] @ [256 x 768], fused roll+window gather.
// Output scattered into per-(window,head) bf16 q/k/v tiles; q pre-scaled.
// ---------------------------------------------------------------------------
__device__ __forceinline__ void store_qkv(float val, int R, int C,
                                          bf16_t* wq, bf16_t* wk, bf16_t* wv) {
  const int part = C >> 8;          // 0=q 1=k 2=v
  const int head = (C & 255) >> 5;
  const int d    = C & 31;
  const int wiG  = R / NTOK;
  const int tok  = R - wiG * NTOK;
  const size_t off = ((size_t)(wiG * HEADS_ + head) * NTOK + tok) * DH + d;
  if (part == 0)      wq[off] = (bf16_t)(val * 0.17677669529663687f); // dh^-0.5
  else if (part == 1) wk[off] = (bf16_t)val;
  else                wv[off] = (bf16_t)val;
}

__global__ __launch_bounds__(256) void qkv_gemm(
    const float* __restrict__ x, const float* __restrict__ qkv_w,
    const float* __restrict__ qkv_b,
    bf16_t* __restrict__ wq, bf16_t* __restrict__ wk, bf16_t* __restrict__ wv) {
  __shared__ bf16_t sA[64][40];    // 64 rows x 32 K (padded)
  __shared__ bf16_t sBT[64][40];   // 64 cols x 32 K (transposed, padded)

  const int mT = blockIdx.x, nT = blockIdx.y;
  const int t = threadIdx.x, wave = t >> 5, lane = t & 31;
  const int wm = (wave & 3) * 16;     // wave row offset in tile
  const int wn = (wave >> 2) * 32;    // wave col offset in tile

  // A gather: thread loads 8 contiguous K for row ar (shift+partition fused)
  const int ar  = t >> 2;
  const int akb = (t & 3) * 8;
  const int R   = mT * 64 + ar;
  const int wiG = R / NTOK;
  const int tok = R - wiG * NTOK;
  const int bb  = wiG >> 8, wix = wiG & 255;
  const int wh  = wix >> 4, ww = wix & 15;
  const int ty  = tok / WSZ, tx = tok - ty * WSZ;
  const int sy  = (wh * WSZ + ty + SHIFT_) % IMG;   // roll(-shift)
  const int sx  = (ww * WSZ + tx + SHIFT_) % IMG;
  const float* arow = x + (((size_t)bb * IMG + sy) * IMG + sx) * CCH + akb;

  // B staging: thread handles K-row pair bkp,bkp+1 x 4 cols -> packed b32 stores
  const int bkp = (t >> 4) * 2;      // 0..30
  const int bcb = (t & 15) * 4;      // 0..60

  v8f acc0 = {}, acc1 = {};
  for (int kt = 0; kt < 8; ++kt) {
    const float4 f0 = *(const float4*)(arow + kt * 32);
    const float4 f1 = *(const float4*)(arow + kt * 32 + 4);
    sA[ar][akb + 0] = (bf16_t)f0.x; sA[ar][akb + 1] = (bf16_t)f0.y;
    sA[ar][akb + 2] = (bf16_t)f0.z; sA[ar][akb + 3] = (bf16_t)f0.w;
    sA[ar][akb + 4] = (bf16_t)f1.x; sA[ar][akb + 5] = (bf16_t)f1.y;
    sA[ar][akb + 6] = (bf16_t)f1.z; sA[ar][akb + 7] = (bf16_t)f1.w;

    const float* brow = qkv_w + (size_t)(kt * 32 + bkp) * 768 + nT * 64 + bcb;
    const float4 g0 = *(const float4*)brow;          // K row bkp
    const float4 g1 = *(const float4*)(brow + 768);  // K row bkp+1
    pack_store(&sBT[bcb + 0][bkp], g0.x, g1.x);
    pack_store(&sBT[bcb + 1][bkp], g0.y, g1.y);
    pack_store(&sBT[bcb + 2][bkp], g0.z, g1.z);
    pack_store(&sBT[bcb + 3][bkp], g0.w, g1.w);

    if (kt < 7) __builtin_prefetch(arow + (kt + 1) * 32, 0, 1); // global_prefetch
    __syncthreads();

    const v16bf a  = frag_a(&sA[wm][0], 40, lane);
    const v16bf b0 = frag_b(&sBT[wn][0], 40, lane);
    const v16bf b1 = frag_b(&sBT[wn + 16][0], 40, lane);
    acc0 = WMMA_BF16(a, b0, acc0);
    acc1 = WMMA_BF16(a, b1, acc1);
    __syncthreads();
  }

  const int half = lane >> 4, cl = lane & 15;
  const int col0 = nT * 64 + wn + cl;
  const float bias0 = qkv_b[col0], bias1 = qkv_b[col0 + 16];
#pragma unroll
  for (int i = 0; i < 8; ++i) {
    const int Rr = mT * 64 + wm + half * 8 + i;
    store_qkv(acc0[i] + bias0, Rr, col0,      wq, wk, wv);
    store_qkv(acc1[i] + bias1, Rr, col0 + 16, wq, wk, wv);
  }
}

// ---------------------------------------------------------------------------
// Kernel 2: per-(window,head) attention. 49 tokens padded to 64.
// S=QK^T (WMMA), +rel-bias (+analytic shift mask), softmax, O=P·V (WMMA).
// ---------------------------------------------------------------------------
__global__ __launch_bounds__(128) void win_attn(
    const bf16_t* __restrict__ wq, const bf16_t* __restrict__ wk,
    const bf16_t* __restrict__ wv, const float* __restrict__ rel_bias,
    bf16_t* __restrict__ wo) {
  __shared__ bf16_t sQ[64][40];
  __shared__ bf16_t sKT[64][40];   // [m][d] == B^T for Q·K^T
  __shared__ bf16_t sVT[32][72];   // [d][m] == B^T for P·V
  __shared__ bf16_t sP[64][72];
  __shared__ float  sS[64][68];
  __shared__ float  sBias[169];
  __shared__ int    sReg[64];

  const int wiH  = blockIdx.x;
  const int wi   = wiH >> 3;
  const int head = wiH & 7;
  const int wix  = wi & 255;
  const int wh   = wix >> 4, ww = wix & 15;
  const int t = threadIdx.x, wave = t >> 5, lane = t & 31;

  const bf16_t* qb = wq + (size_t)wiH * NTOK * DH;
  const bf16_t* kb = wk + (size_t)wiH * NTOK * DH;
  const bf16_t* vb = wv + (size_t)wiH * NTOK * DH;

  for (int idx = t; idx < 64 * 32; idx += 128) {
    const int r = idx >> 5, d = idx & 31;
    bf16_t zq = (bf16_t)0.f, zk = zq;
    if (r < NTOK) { zq = qb[r * 32 + d]; zk = kb[r * 32 + d]; }
    sQ[r][d] = zq; sKT[r][d] = zk;
  }
  // V transpose staged as packed token-pairs -> ds_store_b32
  for (int idx = t; idx < 32 * 32; idx += 128) {
    const int d  = idx & 31;
    const int r  = (idx >> 5) * 2;                 // 0..62, even
    const float lo = (r     < NTOK) ? (float)vb[r * 32 + d]      : 0.f;
    const float hi = (r + 1 < NTOK) ? (float)vb[r * 32 + 32 + d] : 0.f;
    pack_store(&sVT[d][r], lo, hi);
  }
  for (int idx = t; idx < 169; idx += 128) sBias[idx] = rel_bias[idx * HEADS_ + head];
  if (t < 64) {
    int rg = 0;
    if (t < NTOK) {
      const int ty = t / WSZ, tx = t - ty * WSZ;
      const int yw = wh * WSZ + ty, xw = ww * WSZ + tx;
      const int ry = (yw < IMG - WSZ) ? 0 : ((yw < IMG - SHIFT_) ? 1 : 2);
      const int rx = (xw < IMG - WSZ) ? 0 : ((xw < IMG - SHIFT_) ? 1 : 2);
      rg = ry * 3 + rx;
    }
    sReg[t] = rg;
  }
  __syncthreads();

  // S = Q K^T (wave handles a 16-row strip; 4 col tiles)
  {
    const v16bf a = frag_a(&sQ[wave * 16][0], 40, lane);
    const int half = lane >> 4, cl = lane & 15;
#pragma unroll
    for (int cs = 0; cs < 4; ++cs) {
      const v16bf b = frag_b(&sKT[cs * 16][0], 40, lane);
      v8f s = {};
      s = WMMA_BF16(a, b, s);
#pragma unroll
      for (int i = 0; i < 8; ++i) {
        const int n = wave * 16 + half * 8 + i;
        const int m = cs * 16 + cl;
        float v = s[i];
        if (n < NTOK && m < NTOK) {
          const int dy = n / WSZ - m / WSZ + (WSZ - 1);
          const int dx = n % WSZ - m % WSZ + (WSZ - 1);
          v += sBias[dy * (2 * WSZ - 1) + dx];
          if (sReg[n] != sReg[m]) v -= 100.0f;   // shifted-window mask
        }
        sS[n][m] = v;
      }
    }
  }
  __syncthreads();

  // Row softmax -> P (bf16), zero-padded to 64x64
  if (t < 64) {
    if (t < NTOK) {
      float mx = -3.0e38f;
      for (int m = 0; m < NTOK; ++m) mx = fmaxf(mx, sS[t][m]);
      float sum = 0.f;
      for (int m = 0; m < NTOK; ++m) { const float e = __expf(sS[t][m] - mx); sS[t][m] = e; sum += e; }
      const float inv = 1.0f / sum;
      for (int m = 0; m < NTOK; ++m) sP[t][m] = (bf16_t)(sS[t][m] * inv);
      for (int m = NTOK; m < 64; ++m) sP[t][m] = (bf16_t)0.f;
    } else {
      for (int m = 0; m < 64; ++m) sP[t][m] = (bf16_t)0.f;
    }
  }
  __syncthreads();

  // O = P V  (K = 64 -> two K-steps of 32)
  v8f o0 = {}, o1 = {};
#pragma unroll
  for (int ks = 0; ks < 2; ++ks) {
    const v16bf a  = frag_a(&sP[wave * 16][ks * 32], 72, lane);
    const v16bf b0 = frag_b(&sVT[0][ks * 32], 72, lane);
    const v16bf b1 = frag_b(&sVT[16][ks * 32], 72, lane);
    o0 = WMMA_BF16(a, b0, o0);
    o1 = WMMA_BF16(a, b1, o1);
  }
  const int half = lane >> 4, cl = lane & 15;
#pragma unroll
  for (int i = 0; i < 8; ++i) {
    const int n = wave * 16 + half * 8 + i;
    if (n < NTOK) {
      const size_t off = ((size_t)wi * NTOK + n) * CCH + head * DH;
      wo[off + cl]      = (bf16_t)o0[i];
      wo[off + cl + 16] = (bf16_t)o1[i];
    }
  }
}

// ---------------------------------------------------------------------------
// Kernel 3: proj GEMM [100352 x 256] @ [256 x 256] + bias,
// fused window-merge + reverse cyclic shift scatter, fp32 output.
// ---------------------------------------------------------------------------
__global__ __launch_bounds__(256) void proj_gemm(
    const bf16_t* __restrict__ wo, const float* __restrict__ proj_w,
    const float* __restrict__ proj_b, float* __restrict__ out) {
  __shared__ bf16_t sA[64][40];
  __shared__ bf16_t sBT[64][40];

  const int mT = blockIdx.x, nT = blockIdx.y;
  const int t = threadIdx.x, wave = t >> 5, lane = t & 31;
  const int wm = (wave & 3) * 16, wn = (wave >> 2) * 32;

  const int ar = t >> 2, akb = (t & 3) * 8;
  const bf16_t* arow = wo + (size_t)(mT * 64 + ar) * CCH + akb;
  const int bkp = (t >> 4) * 2;      // K-row pair
  const int bcb = (t & 15) * 4;      // col base

  v8f acc0 = {}, acc1 = {};
  for (int kt = 0; kt < 8; ++kt) {
    const bf16_t* ap = arow + kt * 32;
#pragma unroll
    for (int i = 0; i < 8; ++i) sA[ar][akb + i] = ap[i];

    const float* brow = proj_w + (size_t)(kt * 32 + bkp) * CCH + nT * 64 + bcb;
    const float4 g0 = *(const float4*)brow;
    const float4 g1 = *(const float4*)(brow + CCH);
    pack_store(&sBT[bcb + 0][bkp], g0.x, g1.x);
    pack_store(&sBT[bcb + 1][bkp], g0.y, g1.y);
    pack_store(&sBT[bcb + 2][bkp], g0.z, g1.z);
    pack_store(&sBT[bcb + 3][bkp], g0.w, g1.w);

    if (kt < 7) __builtin_prefetch(arow + (kt + 1) * 32, 0, 1);
    __syncthreads();

    const v16bf a  = frag_a(&sA[wm][0], 40, lane);
    const v16bf b0 = frag_b(&sBT[wn][0], 40, lane);
    const v16bf b1 = frag_b(&sBT[wn + 16][0], 40, lane);
    acc0 = WMMA_BF16(a, b0, acc0);
    acc1 = WMMA_BF16(a, b1, acc1);
    __syncthreads();
  }

  const int half = lane >> 4, cl = lane & 15;
  const int col0 = nT * 64 + wn + cl;
  const float pb0 = proj_b[col0], pb1 = proj_b[col0 + 16];
#pragma unroll
  for (int i = 0; i < 8; ++i) {
    const int R   = mT * 64 + wm + half * 8 + i;
    const int wiG = R / NTOK;
    const int tok = R - wiG * NTOK;
    const int bb  = wiG >> 8, wix = wiG & 255;
    const int wh  = wix >> 4, ww = wix & 15;
    const int ty  = tok / WSZ, tx = tok - ty * WSZ;
    const int oy  = (wh * WSZ + ty + SHIFT_) % IMG;   // reverse roll(+shift)
    const int ox  = (ww * WSZ + tx + SHIFT_) % IMG;
    const size_t off = (((size_t)bb * IMG + oy) * IMG + ox) * CCH;
    out[off + col0]      = acc0[i] + pb0;
    out[off + col0 + 16] = acc1[i] + pb1;
  }
}

// ---------------------------------------------------------------------------
extern "C" void kernel_launch(void* const* d_in, const int* in_sizes, int n_in,
                              void* d_out, int out_size, void* d_ws, size_t ws_size,
                              hipStream_t stream) {
  (void)in_sizes; (void)n_in; (void)out_size; (void)ws_size;
  const float* x      = (const float*)d_in[0];
  const float* qkv_w  = (const float*)d_in[1];
  const float* qkv_b  = (const float*)d_in[2];
  const float* proj_w = (const float*)d_in[3];
  const float* proj_b = (const float*)d_in[4];
  const float* rel_b  = (const float*)d_in[5];
  float* out = (float*)d_out;

  const size_t seg = (size_t)NWIN_TOT * NTOK * CCH;  // 25,690,112 elems
  bf16_t* wq = (bf16_t*)d_ws;
  bf16_t* wk = wq + seg;
  bf16_t* wv = wk + seg;
  bf16_t* wo = wv + seg;                              // ~205 MB total scratch

  dim3 g1(MROWS / 64, 768 / 64);
  qkv_gemm<<<g1, 256, 0, stream>>>(x, qkv_w, qkv_b, wq, wk, wv);

  win_attn<<<NWIN_TOT * HEADS_, 128, 0, stream>>>(wq, wk, wv, rel_b, wo);

  dim3 g3(MROWS / 64, CCH / 64);
  proj_gemm<<<g3, 256, 0, stream>>>(wo, proj_w, proj_b, out);
}
